// LSMultiheadAttentionLayer_48266842472688
// MI455X (gfx1250) — compile-verified, matrix operand order checked by reading, not run on hardware
//
#include <hip/hip_runtime.h>
#include <hip/hip_bf16.h>

// ---------------------------------------------------------------------------
// LightSeq-style pre-LN multihead attention layer for MI455X (gfx1250).
// fp32 math for LN/softmax/accum; bf16 operands into v_wmma_f32_16x16x32_bf16.
// GEMMs: 4-wave workgroup, 64x64 tile, A-tile double-buffered into LDS via
// GLOBAL_LOAD_ASYNC_TO_LDS_B128 (ASYNCcnt), B streamed from L2 with prefetch.
// Attention: 1 wave / 32 q-rows, flash-style online softmax, launch_bounds
// raised so the ~250-VGPR working set stays spill-free (wave32 allows 1024).
// B=4, S=2048, H=1024, NH=16, HD=64.
// ---------------------------------------------------------------------------

#define B_   4
#define S_   2048
#define H_   1024
#define NH_  16
#define HD_  64
#define BS_  (B_ * S_)

typedef __attribute__((ext_vector_type(16))) __bf16        v16bf;
typedef __attribute__((ext_vector_type(8)))  float         v8f;
typedef __attribute__((ext_vector_type(4)))  unsigned int  v4u;

struct Frag32B { v4u lo, hi; };

static __device__ __forceinline__ unsigned short f2bfu(float f) {
    unsigned int u = __builtin_bit_cast(unsigned int, f);
    unsigned int r = u + 0x7FFFu + ((u >> 16) & 1u);   // round-to-nearest-even
    return (unsigned short)(r >> 16);
}

// Load one 16x32 bf16 A/B fragment for this lane. `row` points at the start of
// the (stride-aligned) 32-wide K slice of this lane's row/column.
// Per ISA 7.12.2: elements 0..7 = K[8*half .. 8*half+7],
//                 elements 8..15 = K[16+8*half .. 23+8*half].
static __device__ __forceinline__ v16bf load_frag(const unsigned short* row, int half) {
    Frag32B f;
    f.lo = *(const v4u*)(row + 8 * half);
    f.hi = *(const v4u*)(row + 16 + 8 * half);
    return __builtin_bit_cast(v16bf, f);
}

static __device__ __forceinline__ v8f wmma_bf16(v16bf a, v16bf b, v8f c) {
    return __builtin_amdgcn_wmma_f32_16x16x32_bf16(false, a, false, b, (short)0, c,
                                                   false, false);
}

static __device__ __forceinline__ float hmax16(float v) {
    v = fmaxf(v, __shfl_xor(v, 1, 32));
    v = fmaxf(v, __shfl_xor(v, 2, 32));
    v = fmaxf(v, __shfl_xor(v, 4, 32));
    v = fmaxf(v, __shfl_xor(v, 8, 32));
    return v;
}
static __device__ __forceinline__ float hsum16(float v) {
    v += __shfl_xor(v, 1, 32);
    v += __shfl_xor(v, 2, 32);
    v += __shfl_xor(v, 4, 32);
    v += __shfl_xor(v, 8, 32);
    return v;
}

// Issue this thread's two async 16B chunks of the 64x32 bf16 A-tile into LDS.
// Chunk c (0..255): row = c>>2, 8-elem offset = (c&3)*8; thread tid owns
// chunks 2*tid and 2*tid+1 (contiguous in LDS). Tracked by ASYNCcnt.
static __device__ __forceinline__ void async_copy_tileA(const unsigned short* A,
                                                        int m0, int kb,
                                                        unsigned short* buf, int tid) {
    int c0 = tid * 2;
    const unsigned short* g0 = A + (size_t)(m0 + (c0 >> 2)) * H_ + kb + (c0 & 3) * 8;
    const unsigned short* g1 = A + (size_t)(m0 + ((c0 + 1) >> 2)) * H_ + kb + ((c0 + 1) & 3) * 8;
    unsigned la0 = (unsigned)(size_t)buf + (unsigned)c0 * 16u;
    unsigned la1 = la0 + 16u;
    unsigned long long ga0 = (unsigned long long)(size_t)g0;
    unsigned long long ga1 = (unsigned long long)(size_t)g1;
    asm volatile("global_load_async_to_lds_b128 %0, %2, off\n\t"
                 "global_load_async_to_lds_b128 %1, %3, off"
                 :: "v"(la0), "v"(la1), "v"(ga0), "v"(ga1)
                 : "memory");
}

// ---------------------------------------------------------------------------
// Kernel 1: cast fp32 weights -> bf16 (qkv_w [3H,H] and out_w [H,H], row-major;
// rows of W are B-fragment columns, so no transpose is needed).
// ---------------------------------------------------------------------------
__global__ void cvt_weights_kernel(const float* __restrict__ qkv_w,
                                   const float* __restrict__ out_w,
                                   unsigned short* __restrict__ qkv_wb,
                                   unsigned short* __restrict__ out_wb) {
    int idx = blockIdx.x * 256 + threadIdx.x;
    const int NQ = 3 * H_ * H_;
    if (idx < NQ) {
        qkv_wb[idx] = f2bfu(qkv_w[idx]);
    } else {
        int j = idx - NQ;
        out_wb[j] = f2bfu(out_w[j]);
    }
}

// ---------------------------------------------------------------------------
// Kernel 2: pre-LayerNorm, fp32 in -> bf16 out (GEMM A operand).
// ---------------------------------------------------------------------------
__global__ void ln_kernel(const float* __restrict__ x, const float* __restrict__ g,
                          const float* __restrict__ bta,
                          unsigned short* __restrict__ yb) {
    __shared__ float s1[256], s2[256];
    int t = blockIdx.x, tid = threadIdx.x;
    const float* xr = x + (size_t)t * H_;
    float a = 0.f, b2 = 0.f;
    for (int i = tid; i < H_; i += 256) { float v = xr[i]; a += v; b2 += v * v; }
    s1[tid] = a; s2[tid] = b2; __syncthreads();
    for (int o = 128; o > 0; o >>= 1) {
        if (tid < o) { s1[tid] += s1[tid + o]; s2[tid] += s2[tid + o]; }
        __syncthreads();
    }
    float mean = s1[0] * (1.f / H_);
    float var  = s2[0] * (1.f / H_) - mean * mean;
    float rstd = rsqrtf(var + 1e-5f);
    unsigned short* yr = yb + (size_t)t * H_;
    for (int i = tid; i < H_; i += 256)
        yr[i] = f2bfu((xr[i] - mean) * rstd * g[i] + bta[i]);
}

// ---------------------------------------------------------------------------
// Kernels 3/5: 64x64-tile GEMM, 4 waves per block. A-tile (64x32 bf16, shared
// by all waves) double-buffered in LDS via async-to-LDS DMA; each wave owns a
// 16-column slice of B read straight from global (L2-resident weights).
// MODE 0: QKV projection epilogue (+bias, scatter Q/K -> [B,NH,S,HD],
//         V -> [B,NH,HD,S], bf16).
// MODE 1: out-projection epilogue (+bias, +residual, fp32 to d_out).
// ---------------------------------------------------------------------------
template <int MODE>
__global__ void __launch_bounds__(128, 1)
gemm64_kernel(const unsigned short* __restrict__ Ab,
              const unsigned short* __restrict__ Wb,
              const float* __restrict__ bias,
              unsigned short* __restrict__ Qb,
              unsigned short* __restrict__ Kb,
              unsigned short* __restrict__ VTb,
              const float* __restrict__ xres,
              float* __restrict__ outp) {
    __shared__ unsigned short shA[2][64 * 32];
    const int tid = threadIdx.x;
    const int w = tid >> 5, lane = tid & 31;
    const int half = lane >> 4, l15 = lane & 15;
    const int m0 = blockIdx.y * 64;
    const int n0 = blockIdx.x * 64;

    const unsigned short* br = Wb + (size_t)(n0 + w * 16 + l15) * H_;

    v8f acc[4] = {};

    // prologue: start DMA of first A-tile
    async_copy_tileA(Ab, m0, 0, &shA[0][0], tid);

#pragma unroll 1
    for (int kb = 0; kb < H_; kb += 32) {
        const int cur = (kb >> 5) & 1;
        if (kb + 32 < H_) {
            async_copy_tileA(Ab, m0, kb + 32, &shA[cur ^ 1][0], tid);
            __builtin_prefetch(br + kb + 160, 0, 1);
            asm volatile("s_wait_asynccnt 0x2" ::: "memory");  // current tile done
        } else {
            asm volatile("s_wait_asynccnt 0x0" ::: "memory");
        }
        __syncthreads();  // A-tile visible to all 4 waves

        v16bf bfr = load_frag(br + kb, half);
        v16bf a[4];
#pragma unroll
        for (int mi = 0; mi < 4; ++mi)
            a[mi] = load_frag(&shA[cur][(mi * 16 + l15) * 32], half);
#pragma unroll
        for (int mi = 0; mi < 4; ++mi)
            acc[mi] = wmma_bf16(a[mi], bfr, acc[mi]);

        __syncthreads();  // protect buffer before it is DMA-overwritten
    }

    const int e = n0 + w * 16 + l15;   // output feature
    const float bv = bias[e];
    if (MODE == 0) {
        int sec = e >> 10;             // 0=q 1=k 2=v
        int ein = e & (H_ - 1);
        int h = ein >> 6, d = ein & 63;
#pragma unroll
        for (int mi = 0; mi < 4; ++mi)
#pragma unroll
            for (int r = 0; r < 8; ++r) {
                int t = m0 + mi * 16 + 8 * half + r;   // token index
                int bb = t >> 11, s = t & (S_ - 1);
                unsigned short v = f2bfu(acc[mi][r] + bv);
                if (sec == 0)      Qb[(((bb * NH_ + h) * S_ + s) << 6) + d] = v;
                else if (sec == 1) Kb[(((bb * NH_ + h) * S_ + s) << 6) + d] = v;
                else               VTb[(((bb * NH_ + h) * HD_ + d) << 11) + s] = v;
            }
    } else {
#pragma unroll
        for (int mi = 0; mi < 4; ++mi)
#pragma unroll
            for (int r = 0; r < 8; ++r) {
                int t = m0 + mi * 16 + 8 * half + r;
                size_t idx = (size_t)t * H_ + e;
                outp[idx] = acc[mi][r] + bv + xres[idx];
            }
    }
}

// ---------------------------------------------------------------------------
// Kernel 4: flash-style attention. One wave handles (b, head, 32 q-rows),
// streaming kv in chunks of 32 with online softmax. P goes through LDS to be
// re-read in WMMA A-fragment layout. ctx -> bf16 [B,S,H].
// __launch_bounds__(32, 1): lift the VGPR cap so the ~250-VGPR working set
// (Q/K/P/V frags + 8 ctx accums + row state) stays spill-free.
// ---------------------------------------------------------------------------
__global__ void __launch_bounds__(32, 1)
attn_kernel(const unsigned short* __restrict__ Qb,
            const unsigned short* __restrict__ Kb,
            const unsigned short* __restrict__ VTb,
            const int* __restrict__ maskp,
            unsigned short* __restrict__ ctxb) {
    __shared__ unsigned short pl[32 * 32];
    const int lane = threadIdx.x, half = lane >> 4, l15 = lane & 15;
    const int bid = blockIdx.x;
    const int qt = bid & 63, hh = (bid >> 6) & 15, bb = bid >> 10;

    const unsigned short* Qh  = Qb  + ((size_t)(bb * NH_ + hh) * S_) * HD_;
    const unsigned short* Kh  = Kb  + ((size_t)(bb * NH_ + hh) * S_) * HD_;
    const unsigned short* VTh = VTb + ((size_t)(bb * NH_ + hh) * HD_) * S_;
    const int* mptr = maskp + bb * S_;

    v16bf aq[2][2];
#pragma unroll
    for (int mi = 0; mi < 2; ++mi)
#pragma unroll
        for (int kk = 0; kk < 2; ++kk)
            aq[mi][kk] = load_frag(Qh + (size_t)(qt * 32 + mi * 16 + l15) * HD_ + kk * 32, half);

    float mrow[16], lrow[16];
#pragma unroll
    for (int i = 0; i < 16; ++i) { mrow[i] = -3.0e38f; lrow[i] = 0.f; }
    v8f cacc[2][4] = {};

    for (int kv0 = 0; kv0 < S_; kv0 += 32) {
        // scores: S = (Q K^T) * 0.125 + mask * -1e8
        v16bf bk[2][2];
#pragma unroll
        for (int ni = 0; ni < 2; ++ni)
#pragma unroll
            for (int kk = 0; kk < 2; ++kk)
                bk[ni][kk] = load_frag(Kh + (size_t)(kv0 + ni * 16 + l15) * HD_ + kk * 32, half);
        v8f sf[2][2] = {};
#pragma unroll
        for (int mi = 0; mi < 2; ++mi)
#pragma unroll
            for (int ni = 0; ni < 2; ++ni)
#pragma unroll
                for (int kk = 0; kk < 2; ++kk)
                    sf[mi][ni] = wmma_bf16(aq[mi][kk], bk[ni][kk], sf[mi][ni]);

        float mb[2];
#pragma unroll
        for (int ni = 0; ni < 2; ++ni)
            mb[ni] = (float)mptr[kv0 + ni * 16 + l15] * -1e8f;

#pragma unroll
        for (int mi = 0; mi < 2; ++mi)
#pragma unroll
            for (int ni = 0; ni < 2; ++ni)
#pragma unroll
                for (int r = 0; r < 8; ++r)
                    sf[mi][ni][r] = sf[mi][ni][r] * 0.125f + mb[ni];

        // online softmax per row (rows of a lane: m = mi*16 + 8*half + r)
#pragma unroll
        for (int mi = 0; mi < 2; ++mi)
#pragma unroll
            for (int r = 0; r < 8; ++r) {
                int ri = mi * 8 + r;
                float cm = hmax16(fmaxf(sf[mi][0][r], sf[mi][1][r]));
                float mo = mrow[ri];
                float mn = fmaxf(mo, cm);
                float resc = __expf(mo - mn);
                mrow[ri] = mn;
                float p0 = __expf(sf[mi][0][r] - mn);
                float p1 = __expf(sf[mi][1][r] - mn);
                sf[mi][0][r] = p0; sf[mi][1][r] = p1;
                lrow[ri] = lrow[ri] * resc + hsum16(p0 + p1);
#pragma unroll
                for (int nf = 0; nf < 4; ++nf) cacc[mi][nf][r] *= resc;
            }

        __syncthreads();
#pragma unroll
        for (int mi = 0; mi < 2; ++mi)
#pragma unroll
            for (int ni = 0; ni < 2; ++ni)
#pragma unroll
                for (int r = 0; r < 8; ++r) {
                    int m = mi * 16 + 8 * half + r, c = ni * 16 + l15;
                    pl[m * 32 + c] = f2bfu(sf[mi][ni][r]);
                }
        __syncthreads();

        v16bf ap[2], bv[4];
#pragma unroll
        for (int mi = 0; mi < 2; ++mi)
            ap[mi] = load_frag(&pl[(mi * 16 + l15) * 32], half);
#pragma unroll
        for (int nf = 0; nf < 4; ++nf)
            bv[nf] = load_frag(VTh + (size_t)(nf * 16 + l15) * S_ + kv0, half);
#pragma unroll
        for (int mi = 0; mi < 2; ++mi)
#pragma unroll
            for (int nf = 0; nf < 4; ++nf)
                cacc[mi][nf] = wmma_bf16(ap[mi], bv[nf], cacc[mi][nf]);
    }

#pragma unroll
    for (int mi = 0; mi < 2; ++mi)
#pragma unroll
        for (int r = 0; r < 8; ++r) {
            int ri = mi * 8 + r;
            float inv = 1.0f / lrow[ri];
            int s = qt * 32 + mi * 16 + 8 * half + r;
#pragma unroll
            for (int nf = 0; nf < 4; ++nf) {
                int col = hh * HD_ + nf * 16 + l15;
                ctxb[((size_t)(bb * S_ + s) * H_) + col] = f2bfu(cacc[mi][nf][r] * inv);
            }
        }
}

// ---------------------------------------------------------------------------
extern "C" void kernel_launch(void* const* d_in, const int* in_sizes, int n_in,
                              void* d_out, int out_size, void* d_ws, size_t ws_size,
                              hipStream_t stream) {
    (void)in_sizes; (void)n_in; (void)out_size; (void)ws_size;
    const float* x       = (const float*)d_in[0];
    const int*   mask    = (const int*)d_in[1];
    const float* qkv_w   = (const float*)d_in[2];
    const float* qkv_b   = (const float*)d_in[3];
    const float* out_w   = (const float*)d_in[4];
    const float* out_b   = (const float*)d_in[5];
    const float* ln_g    = (const float*)d_in[6];
    const float* ln_b    = (const float*)d_in[7];
    float* out = (float*)d_out;

    // workspace layout (bf16 buffers), ~88 MB total
    char* w = (char*)d_ws;
    unsigned short* yb     = (unsigned short*)w;                 w += (size_t)BS_ * H_ * 2;       // 16 MB
    unsigned short* qkv_wb = (unsigned short*)w;                 w += (size_t)3 * H_ * H_ * 2;    //  6 MB
    unsigned short* out_wb = (unsigned short*)w;                 w += (size_t)H_ * H_ * 2;        //  2 MB
    unsigned short* Qb     = (unsigned short*)w;                 w += (size_t)BS_ * H_ * 2;       // 16 MB
    unsigned short* Kb     = (unsigned short*)w;                 w += (size_t)BS_ * H_ * 2;       // 16 MB
    unsigned short* VTb    = (unsigned short*)w;                 w += (size_t)BS_ * H_ * 2;       // 16 MB
    unsigned short* ctxb   = (unsigned short*)w;                                                  // 16 MB

    cvt_weights_kernel<<<(3 * H_ * H_ + H_ * H_) / 256, 256, 0, stream>>>(
        qkv_w, out_w, qkv_wb, out_wb);
    ln_kernel<<<BS_, 256, 0, stream>>>(x, ln_g, ln_b, yb);
    gemm64_kernel<0><<<dim3(3 * H_ / 64, BS_ / 64), 128, 0, stream>>>(
        yb, qkv_wb, qkv_b, Qb, Kb, VTb, nullptr, nullptr);
    attn_kernel<<<B_ * NH_ * (S_ / 32), 32, 0, stream>>>(Qb, Kb, VTb, mask, ctxb);
    gemm64_kernel<1><<<dim3(H_ / 64, BS_ / 64), 128, 0, stream>>>(
        ctxb, out_wb, out_b, nullptr, nullptr, nullptr, x, out);
}
